// Contextual_Attention_18442589569438
// MI455X (gfx1250) — matrix operand order repaired and checked by reading, not compiled, not run on hardware
//
#include <hip/hip_runtime.h>
#include <hip/hip_bf16.h>

// ---------------------------------------------------------------------------
// Contextual attention (DeepFill) for MI455X / gfx1250, wave32 + WMMA bf16.
// B=2, C=128, H=W=128, RATE=2 -> fh=fw=bh=bw=64, L=P=4096, K1=C*9=1152,
// N2=C*16=2048. Two bf16 WMMA GEMMs dominate (~215 GFLOP total).
// K-dim of GEMM1 reordered as (g=ky*3+kx, c) so each 32-wide K-chunk has a
// uniform (ky,kx): no per-element integer division in the hot loop.
// Both GEMMs use double-buffered LDS tiles (one barrier per K-step).
// ---------------------------------------------------------------------------

typedef __attribute__((ext_vector_type(16))) __bf16 v16bf;
typedef __attribute__((ext_vector_type(8)))  float  v8f;

struct U8 { unsigned int u[8]; };

__device__ __forceinline__ v16bf as_bf16(const U8 &x) {
  return __builtin_bit_cast(v16bf, x);
}

// round-to-nearest-even f32 -> bf16
__device__ __forceinline__ unsigned short f2bf(float f) {
  unsigned int u = __float_as_uint(f);
  u += 0x7FFFu + ((u >> 16) & 1u);
  return (unsigned short)(u >> 16);
}

__device__ __forceinline__ int swapi(int i) { return ((i & 63) << 6) | (i >> 6); }

#define LDSW 34   // LDS row pitch in bf16 elems (17 dwords, odd -> no bank conflicts)

// ---------------------------------------------------------------------------
// Stage 0: downsample f,b (::2,::2) and mask.
// ---------------------------------------------------------------------------
__global__ __launch_bounds__(256)
void downsample_kernel(const float* __restrict__ f, const float* __restrict__ b,
                       const float* __restrict__ mask,
                       float* __restrict__ fd, float* __restrict__ bd,
                       float* __restrict__ md)
{
  int idx = blockIdx.x * 256 + threadIdx.x;        // 128*64*64 = 524288
  if (idx >= 128 * 64 * 64) return;
  int x = idx & 63, y = (idx >> 6) & 63, c = idx >> 12;
  size_t src = ((size_t)c * 128 + 2 * y) * 128 + 2 * x;
  fd[idx] = f[src];
  bd[idx] = b[src];
  if (c == 0) md[y * 64 + x] = mask[(size_t)(2 * y) * 128 + 2 * x];
}

// ---------------------------------------------------------------------------
// Stage 1: per-background-patch inverse L2 norm + mask gate mm[l].
// ---------------------------------------------------------------------------
__global__ __launch_bounds__(256)
void norm_mm_kernel(const float* __restrict__ bd, const float* __restrict__ md,
                    float* __restrict__ invn, float* __restrict__ mm)
{
  int l = blockIdx.x * 256 + threadIdx.x;          // 4096
  if (l >= 4096) return;
  int ly = l >> 6, lx = l & 63;
  float s = 0.0f;
  for (int c = 0; c < 128; ++c)
    for (int ky = 0; ky < 3; ++ky)
      for (int kx = 0; kx < 3; ++kx) {
        int y = ly + ky - 1, x = lx + kx - 1;
        if ((unsigned)y < 64u && (unsigned)x < 64u) {
          float v = bd[((c << 6) + y) * 64 + x];
          s += v * v;
        }
      }
  invn[l] = 1.0f / fmaxf(sqrtf(s), 1.0e-4f);
  float msum = 0.0f;
  for (int ky = 0; ky < 3; ++ky)
    for (int kx = 0; kx < 3; ++kx) {
      int y = ly + ky - 1, x = lx + kx - 1;
      if ((unsigned)y < 64u && (unsigned)x < 64u) msum += md[y * 64 + x];
    }
  mm[l] = (msum == 0.0f) ? 1.0f : 0.0f;
}

// ---------------------------------------------------------------------------
// Stage 2: GEMM1 scores S[l,p] = invn[l] * <bd_patch3x3[l], fd_im2col[p]>
// M=4096(l) N=4096(p) K=1152 ordered (g,c). bf16 WMMA 16x16x32, f32 acc.
// 256 threads = 8 waves; 128x128 tile; wave = 32x64 (2x4 WMMA subtiles).
// ---------------------------------------------------------------------------
__global__ __launch_bounds__(256)
void gemm1_scores(const float* __restrict__ bd, const float* __restrict__ fd,
                  const float* __restrict__ invn, float* __restrict__ S)
{
  __shared__ unsigned short shA[2][128 * LDSW];
  __shared__ unsigned short shB[2][128 * LDSW];

  const int tid  = threadIdx.x;
  const int lane = tid & 31;
  const int wave = tid >> 5;
  const int wm   = wave >> 1;      // 0..3  : 32 rows each
  const int wn   = wave & 1;       // 0..1  : 64 cols each
  const int half = lane >> 4;
  const int l16  = lane & 15;

  const int p0 = blockIdx.x * 128;
  const int l0 = blockIdx.y * 128;

  v8f acc[2][4];
  #pragma unroll
  for (int ms = 0; ms < 2; ++ms)
    #pragma unroll
    for (int ns = 0; ns < 4; ++ns)
      #pragma unroll
      for (int j = 0; j < 8; ++j) acc[ms][ns][j] = 0.0f;

  // k = g*128 + c ; one 32-chunk never crosses a g boundary (128 % 32 == 0).
  auto load_tiles = [&](int buf, int kk) {
    int g  = kk >> 7;              // uniform per K-step
    int cb = kk & 127;
    int ky = g / 3;                // scalar (kk uniform)
    int kx = g - ky * 3;
    #pragma unroll
    for (int it = 0; it < 8; ++it) {
      int pi  = it * 256 + tid;    // 2048 bf16-pairs, row-fastest
      int row = pi & 127;
      int kp  = (pi >> 7) << 1;
      int c0  = cb + kp;
      // A tile: background 3x3 patches (gathered from bd, zero-padded)
      {
        int l = l0 + row, ly = l >> 6, lx = l & 63;
        int y = ly + ky - 1, x = lx + kx - 1;
        bool ok = ((unsigned)y < 64u) && ((unsigned)x < 64u);
        int base = c0 * 4096 + y * 64 + x;
        float v0 = ok ? bd[base]        : 0.0f;
        float v1 = ok ? bd[base + 4096] : 0.0f;
        *(unsigned int*)&shA[buf][row * LDSW + kp] =
            (unsigned int)f2bf(v0) | ((unsigned int)f2bf(v1) << 16);
      }
      // B tile: foreground im2col, stored [p][k] (K-pairs contiguous)
      {
        int p = p0 + row, fy = p >> 6, fx = p & 63;
        int y = fy + ky - 1, x = fx + kx - 1;
        bool ok = ((unsigned)y < 64u) && ((unsigned)x < 64u);
        int base = c0 * 4096 + y * 64 + x;
        float v0 = ok ? fd[base]        : 0.0f;
        float v1 = ok ? fd[base + 4096] : 0.0f;
        *(unsigned int*)&shB[buf][row * LDSW + kp] =
            (unsigned int)f2bf(v0) | ((unsigned int)f2bf(v1) << 16);
      }
    }
  };

  load_tiles(0, 0);

  #pragma unroll 1
  for (int kk = 0; kk < 1152; kk += 32) {
    const int buf = (kk >> 5) & 1;
    __syncthreads();
    if (kk + 32 < 1152) load_tiles(buf ^ 1, kk + 32);

    U8 bfrag[4];
    #pragma unroll
    for (int ns = 0; ns < 4; ++ns) {
      int col = wn * 64 + ns * 16 + l16;       // B frag: N=lane%16, K=(j>>2)*16+half*8+2*(j&3)
      #pragma unroll
      for (int j = 0; j < 8; ++j)
        bfrag[ns].u[j] = *(const unsigned int*)&shB[buf][col * LDSW + (j >> 2) * 16 + half * 8 + 2 * (j & 3)];
    }
    #pragma unroll
    for (int ms = 0; ms < 2; ++ms) {
      U8 afrag;                                // A frag: M=lane%16
      int rowA = wm * 32 + ms * 16 + l16;
      #pragma unroll
      for (int j = 0; j < 4; ++j)              // K = half*8 + 2j
        afrag.u[j] = *(const unsigned int*)&shA[buf][rowA * LDSW + half * 8 + 2 * j];
      #pragma unroll
      for (int j = 0; j < 4; ++j)              // K = 16 + half*8 + 2j
        afrag.u[4 + j] = *(const unsigned int*)&shA[buf][rowA * LDSW + 16 + half * 8 + 2 * j];
      #pragma unroll
      for (int ns = 0; ns < 4; ++ns)
        acc[ms][ns] = __builtin_amdgcn_wmma_f32_16x16x32_bf16(
            false, as_bf16(afrag), false, as_bf16(bfrag[ns]),
            (short)0, acc[ms][ns], false, false);
    }
  }

  // epilogue: scale row by invn and store (D layout: row = j + 8*half)
  #pragma unroll
  for (int ms = 0; ms < 2; ++ms)
    #pragma unroll
    for (int j = 0; j < 8; ++j) {
      int row = l0 + wm * 32 + ms * 16 + (half << 3) + j;
      float sc = invn[row];
      #pragma unroll
      for (int ns = 0; ns < 4; ++ns) {
        int col = p0 + wn * 64 + ns * 16 + l16;
        S[(size_t)row * 4096 + col] = acc[ms][ns][j] * sc;
      }
    }
}

// ---------------------------------------------------------------------------
// Stage 3: fuse — the two eye(3x3) convs on the flattened 4096x4096 matrix
// (including flat-index wrap semantics) as a 9-point gather.
// ---------------------------------------------------------------------------
__global__ __launch_bounds__(256)
void fuse_kernel(const float* __restrict__ S, float* __restrict__ F)
{
  int idx = blockIdx.x * 256 + threadIdx.x;        // 4096*4096
  int p = idx & 4095;
  int l = idx >> 12;
  int sl = swapi(l), sp = swapi(p);
  float acc = 0.0f;
  #pragma unroll
  for (int k2 = 0; k2 < 3; ++k2) {
    int rl = sl + k2 - 1, rp = sp + k2 - 1;
    if ((unsigned)rl < 4096u && (unsigned)rp < 4096u) {
      int a0 = swapi(rl), b0 = swapi(rp);
      #pragma unroll
      for (int k1 = 0; k1 < 3; ++k1) {
        int a = a0 + k1 - 1, b = b0 + k1 - 1;
        if ((unsigned)a < 4096u && (unsigned)b < 4096u)
          acc += S[(size_t)a * 4096 + b];
      }
    }
  }
  F[(size_t)l * 4096 + p] = acc;
}

// ---------------------------------------------------------------------------
// Stage 4: column softmax stats over l (logits = 10 * F * mm[l]).
// 4-way row split per column, LDS reduction. Block = 64 cols x 4 chunks.
// ---------------------------------------------------------------------------
__global__ __launch_bounds__(256)
void softmax_stats(const float* __restrict__ F, const float* __restrict__ mm,
                   float* __restrict__ cmax, float* __restrict__ csum)
{
  __shared__ float red[256];
  int pc = threadIdx.x & 63;                       // column within block
  int tq = threadIdx.x >> 6;                       // row chunk 0..3
  int p  = blockIdx.x * 64 + pc;                   // 64 blocks cover 4096 cols
  int lA = tq * 1024, lB = lA + 1024;

  float m = -3.4e38f;
  for (int l = lA; l < lB; ++l) {
    float v = 10.0f * F[(size_t)l * 4096 + p] * mm[l];
    m = fmaxf(m, v);
  }
  red[threadIdx.x] = m;
  __syncthreads();
  float gm = fmaxf(fmaxf(red[pc], red[pc + 64]),
                   fmaxf(red[pc + 128], red[pc + 192]));
  __syncthreads();

  float s = 0.0f;
  for (int l = lA; l < lB; ++l) {
    float v = 10.0f * F[(size_t)l * 4096 + p] * mm[l];
    s += __expf(v - gm);
  }
  red[threadIdx.x] = s;
  __syncthreads();
  if (tq == 0) {
    cmax[p] = gm;
    csum[p] = red[pc] + red[pc + 64] + red[pc + 128] + red[pc + 192];
  }
}

// ---------------------------------------------------------------------------
// Stage 5: att = mm[l] * softmax, written transposed as bf16: attT[p][l].
// 32x32 LDS tile transpose so both global accesses are coalesced.
// ---------------------------------------------------------------------------
__global__ __launch_bounds__(256)
void att_kernel(const float* __restrict__ F, const float* __restrict__ mm,
                const float* __restrict__ cmax, const float* __restrict__ csum,
                unsigned short* __restrict__ attT)
{
  __shared__ float tile[32][33];
  int tx = threadIdx.x & 31;
  int ty = threadIdx.x >> 5;                       // 0..7
  int p0 = blockIdx.x * 32;
  int l0 = blockIdx.y * 32;
  #pragma unroll
  for (int i = 0; i < 32; i += 8)
    tile[ty + i][tx] = F[(size_t)(l0 + ty + i) * 4096 + p0 + tx];
  __syncthreads();
  #pragma unroll
  for (int i = 0; i < 32; i += 8) {
    int p = p0 + ty + i;
    int l = l0 + tx;
    float v = tile[tx][ty + i];
    float g = mm[l];
    float a = g * __expf(10.0f * v * g - cmax[p]) / csum[p];
    attT[(size_t)p * 4096 + l] = f2bf(a);
  }
}

// ---------------------------------------------------------------------------
// Stage 6: GEMM2 Z[p,n] = sum_l attT[p][l] * b_patch4x4[l][n]
// M=4096(p) N=2048(n=c*16+dy*4+dx) K=4096(l). Same WMMA tiling as GEMM1.
// ---------------------------------------------------------------------------
__global__ __launch_bounds__(256)
void gemm2_paste(const unsigned short* __restrict__ attT,
                 const float* __restrict__ bsrc, float* __restrict__ Z)
{
  __shared__ unsigned short shA[2][128 * LDSW];
  __shared__ unsigned short shB[2][128 * LDSW];

  const int tid  = threadIdx.x;
  const int lane = tid & 31;
  const int wave = tid >> 5;
  const int wm   = wave >> 1;
  const int wn   = wave & 1;
  const int half = lane >> 4;
  const int l16  = lane & 15;

  const int n0 = blockIdx.x * 128;
  const int pB = blockIdx.y * 128;

  v8f acc[2][4];
  #pragma unroll
  for (int ms = 0; ms < 2; ++ms)
    #pragma unroll
    for (int ns = 0; ns < 4; ++ns)
      #pragma unroll
      for (int j = 0; j < 8; ++j) acc[ms][ns][j] = 0.0f;

  auto load_tiles = [&](int buf, int kk) {
    #pragma unroll
    for (int it = 0; it < 8; ++it) {
      int pi  = it * 256 + tid;
      int row = pi >> 4;
      int kp  = (pi & 15) << 1;
      // A tile: attention, already bf16 row-major [p][l] -> straight u32 copy
      *(unsigned int*)&shA[buf][row * LDSW + kp] =
          *(const unsigned int*)(attT + (size_t)(pB + row) * 4096 + kk + kp);
      // B tile: raw 4x4 stride-2 pad-1 patches of b, gathered, stored [n][k]
      {
        int n  = n0 + row;
        int c  = n >> 4;
        int dy = (n >> 2) & 3;
        int dx = n & 3;
        unsigned int pr = 0;
        #pragma unroll
        for (int e = 0; e < 2; ++e) {
          int l  = kk + kp + e;
          int ly = l >> 6, lx = l & 63;
          int y = 2 * ly - 1 + dy, x = 2 * lx - 1 + dx;
          float v = 0.0f;
          if ((unsigned)y < 128u && (unsigned)x < 128u)
            v = bsrc[((c << 7) + y) * 128 + x];
          pr |= (unsigned int)f2bf(v) << (16 * e);
        }
        *(unsigned int*)&shB[buf][row * LDSW + kp] = pr;
      }
    }
  };

  load_tiles(0, 0);

  #pragma unroll 1
  for (int kk = 0; kk < 4096; kk += 32) {
    const int buf = (kk >> 5) & 1;
    __syncthreads();
    if (kk + 32 < 4096) load_tiles(buf ^ 1, kk + 32);

    U8 bfrag[4];
    #pragma unroll
    for (int ns = 0; ns < 4; ++ns) {
      int col = wn * 64 + ns * 16 + l16;
      #pragma unroll
      for (int j = 0; j < 8; ++j)
        bfrag[ns].u[j] = *(const unsigned int*)&shB[buf][col * LDSW + (j >> 2) * 16 + half * 8 + 2 * (j & 3)];
    }
    #pragma unroll
    for (int ms = 0; ms < 2; ++ms) {
      U8 afrag;
      int rowA = wm * 32 + ms * 16 + l16;
      #pragma unroll
      for (int j = 0; j < 4; ++j)
        afrag.u[j] = *(const unsigned int*)&shA[buf][rowA * LDSW + half * 8 + 2 * j];
      #pragma unroll
      for (int j = 0; j < 4; ++j)
        afrag.u[4 + j] = *(const unsigned int*)&shA[buf][rowA * LDSW + 16 + half * 8 + 2 * j];
      #pragma unroll
      for (int ns = 0; ns < 4; ++ns)
        acc[ms][ns] = __builtin_amdgcn_wmma_f32_16x16x32_bf16(
            false, as_bf16(afrag), false, as_bf16(bfrag[ns]),
            (short)0, acc[ms][ns], false, false);
    }
  }

  #pragma unroll
  for (int ms = 0; ms < 2; ++ms)
    #pragma unroll
    for (int j = 0; j < 8; ++j) {
      int row = pB + wm * 32 + ms * 16 + (half << 3) + j;
      #pragma unroll
      for (int ns = 0; ns < 4; ++ns) {
        int col = n0 + wn * 64 + ns * 16 + l16;
        Z[(size_t)row * 2048 + col] = acc[ms][ns][j];
      }
    }
}

// ---------------------------------------------------------------------------
// Stage 7: transposed-conv gather: out[c,u,v] = 0.25 * sum of <=4 Z entries.
// ---------------------------------------------------------------------------
__global__ __launch_bounds__(256)
void scatter_out(const float* __restrict__ Z, float* __restrict__ out)
{
  int idx = blockIdx.x * 256 + threadIdx.x;        // 128*128*128
  if (idx >= 128 * 128 * 128) return;
  int v = idx & 127, u = (idx >> 7) & 127, c = idx >> 14;
  float acc = 0.0f;
  #pragma unroll
  for (int dy = 0; dy < 4; ++dy) {
    int uy = u + 1 - dy;
    if (uy < 0 || (uy & 1)) continue;
    int py = uy >> 1;
    if (py >= 64) continue;
    #pragma unroll
    for (int dx = 0; dx < 4; ++dx) {
      int ux = v + 1 - dx;
      if (ux < 0 || (ux & 1)) continue;
      int px = ux >> 1;
      if (px >= 64) continue;
      acc += Z[(size_t)(py * 64 + px) * 2048 + c * 16 + dy * 4 + dx];
    }
  }
  out[idx] = 0.25f * acc;
}

// ---------------------------------------------------------------------------
extern "C" void kernel_launch(void* const* d_in, const int* in_sizes, int n_in,
                              void* d_out, int out_size, void* d_ws, size_t ws_size,
                              hipStream_t stream)
{
  (void)in_sizes; (void)n_in; (void)out_size; (void)ws_size;
  const float* f    = (const float*)d_in[0];   // (2,128,128,128)
  const float* b    = (const float*)d_in[1];   // (2,128,128,128)
  const float* mask = (const float*)d_in[2];   // (1,1,128,128)
  float* out = (float*)d_out;                  // (2,128,128,128)

  char* ws = (char*)d_ws;
  float* S = (float*)ws;                             // 64 MB scores
  float* F = (float*)(ws + ((size_t)64 << 20));      // 64 MB fused scores
  float* fd   = (float*)(ws + ((size_t)128 << 20));  // 2 MB
  float* bd   = fd + 128 * 64 * 64;                  // 2 MB
  float* md   = bd + 128 * 64 * 64;                  // 16 KB
  float* invn = md + 64 * 64;
  float* mm   = invn + 4096;
  float* cmax = mm + 4096;
  float* csum = cmax + 4096;
  unsigned short* attT = (unsigned short*)S;         // reuse S[0..32MB) as bf16 att
  float* Z = (float*)(ws + ((size_t)32 << 20));      // reuse S[32..64MB)

  const size_t plane = (size_t)128 * 128 * 128;
  for (int i = 0; i < 2; ++i) {
    const float* fi = f + i * plane;
    const float* bi = b + i * plane;
    downsample_kernel<<<2048, 256, 0, stream>>>(fi, bi, mask, fd, bd, md);
    norm_mm_kernel<<<16, 256, 0, stream>>>(bd, md, invn, mm);
    gemm1_scores<<<dim3(32, 32), 256, 0, stream>>>(bd, fd, invn, S);
    fuse_kernel<<<65536, 256, 0, stream>>>(S, F);
    softmax_stats<<<64, 256, 0, stream>>>(F, mm, cmax, csum);
    att_kernel<<<dim3(128, 128), 256, 0, stream>>>(F, mm, cmax, csum, attT);
    gemm2_paste<<<dim3(16, 32), 256, 0, stream>>>(attT, bi, Z);
    scatter_out<<<8192, 256, 0, stream>>>(Z, out + i * plane);
  }
}
